// MultiHeadPerformerAttention_25254407701106
// MI455X (gfx1250) — matrix operand order repaired and checked by reading, not compiled, not run on hardware
//
#include <hip/hip_runtime.h>
#include <hip/hip_bf16.h>

// Performer (FAVOR+) multi-head attention for MI455X (gfx1250, wave32, WMMA).
// All GEMMs use v_wmma_f32_16x16x32_bf16; nonlinear math in f32 VALU.
// bf16->bf16 staging uses GLOBAL_LOAD_ASYNC_TO_LDS_B128 (builtin confirmed
// present in this toolchain), plus global_prefetch_b8 for next-tile prefetch.

#define BB 4
#define LL 4096
#define DM 1024
#define HH 16
#define DEPTHD 64
#define MMF 256
#define NKV 80            // 64 V columns + 1 "ones" column (k_sum) + 15 zero pad
#define NROWS (BB*LL)     // 16384

typedef __attribute__((ext_vector_type(16))) __bf16 v16bf;
typedef __attribute__((ext_vector_type(8)))  float  v8f;

union BFrag { v16bf v; uint4 q[2]; };
union H16  { uint4 q[2]; unsigned short h[16]; };

// ---- optional CDNA5 async global->LDS path (compile-guarded) --------------
#if defined(__has_builtin)
#  if __has_builtin(__builtin_amdgcn_global_load_async_to_lds_b128) && \
      __has_builtin(__builtin_amdgcn_s_wait_asynccnt)
#    define USE_ASYNC_LDS 1
#  endif
#endif
#ifndef USE_ASYNC_LDS
#  define USE_ASYNC_LDS 0
#endif

#if USE_ASYNC_LDS
#define AS_GLOBAL __attribute__((address_space(1)))
#define AS_LDS    __attribute__((address_space(3)))
typedef int v4i_t __attribute__((vector_size(16)));   // matches builtin param type
__device__ inline void async_copy16(const void* g, void* l) {
    __builtin_amdgcn_global_load_async_to_lds_b128(
        (AS_GLOBAL v4i_t*)g, (AS_LDS v4i_t*)l, 0, 0);
}
__device__ inline void async_wait0() { __builtin_amdgcn_s_wait_asynccnt(0); }
#endif

__device__ inline unsigned short bf16_of(float f) {
    unsigned u = __float_as_uint(f);
    u += 0x7FFFu + ((u >> 16) & 1u);           // round-to-nearest-even
    return (unsigned short)(u >> 16);
}
__device__ inline float f_of_bf16(unsigned short h) {
    return __uint_as_float(((unsigned)h) << 16);
}
__device__ inline unsigned pack2(float a, float b) {
    return (unsigned)bf16_of(a) | ((unsigned)bf16_of(b) << 16);
}
// order-preserving float->uint encoding for atomicMax
__device__ inline unsigned enc_max(float f) {
    unsigned u = __float_as_uint(f);
    return (u & 0x80000000u) ? ~u : (u | 0x80000000u);
}
__device__ inline float dec_max(unsigned e) {
    unsigned u = (e & 0x80000000u) ? (e & 0x7FFFFFFFu) : ~e;
    return __uint_as_float(u);
}

__device__ inline v8f wmma_bf16(const BFrag& a, const BFrag& b, v8f c) {
    return __builtin_amdgcn_wmma_f32_16x16x32_bf16(
        false, a.v, false, b.v, (short)0, c, false, false);
}

// ---------------------------------------------------------------------------
// GEMM: C[NROWS,1024] = A[NROWS,1024] @ W[1024,1024] + bias
// WG tile 128x128, 256 threads = 8 waves, each wave 64x32 (4x2 wmma tiles).
// A fp32 (projections) or bf16 (final); C bf16 (q/k/v) or fp32 (output).
// ---------------------------------------------------------------------------
template<bool ABF16, bool OBF16>
__global__ void __launch_bounds__(256)
gemm_bias_kernel(const void* __restrict__ Aptr,
                 const float* __restrict__ W,
                 const float* __restrict__ bias,
                 void* __restrict__ Cptr)
{
    __shared__ __align__(16) unsigned short As[128 * 40];  // [row][k], stride 40 (80B, 16B-div)
    __shared__ __align__(16) unsigned short Bs[128 * 40];  // [col][k] (W transposed)

    const int tid  = threadIdx.x;
    const int lane = tid & 31;
    const int wave = tid >> 5;
    const int rowBase = blockIdx.x * 128;
    const int colBase = blockIdx.y * 128;
    const int wM = wave & 1;       // 0..1 -> 64-row half
    const int wN = wave >> 1;      // 0..3 -> 32-col quarter

    v8f acc[4][2] = {};

    const int stRow = tid >> 1;            // 0..127
    const int stK   = (tid & 1) * 16;      // 0 / 16
    const int wkRow = tid >> 3;            // 0..31
    const int wnP   = (tid & 7) * 16;      // 0..112

    for (int k0 = 0; k0 < DM; k0 += 32) {
        // prefetch next K tiles into cache while we stage/compute this one
        if (k0 + 32 < DM) {
            if (ABF16) {
                const unsigned short* A = (const unsigned short*)Aptr;
                __builtin_prefetch(A + (size_t)(rowBase + stRow) * DM + (k0 + 32) + stK, 0, 3);
            } else {
                const float* A = (const float*)Aptr;
                __builtin_prefetch(A + (size_t)(rowBase + stRow) * DM + (k0 + 32) + stK, 0, 3);
            }
            __builtin_prefetch(W + (size_t)(k0 + 32 + wkRow) * DM + colBase + wnP, 0, 3);
        }
        __syncthreads();
        // ---- stage A tile (128 x 32) as bf16 ----
        if (ABF16) {
            const unsigned short* A = (const unsigned short*)Aptr;
            const unsigned short* src = A + (size_t)(rowBase + stRow) * DM + k0 + stK;
#if USE_ASYNC_LDS
            async_copy16(src,     &As[stRow * 40 + stK]);
            async_copy16(src + 8, &As[stRow * 40 + stK + 8]);
#else
            uint4 a0 = ((const uint4*)src)[0], a1 = ((const uint4*)src)[1];
            *(uint4*)&As[stRow * 40 + stK]     = a0;
            *(uint4*)&As[stRow * 40 + stK + 8] = a1;
#endif
        } else {
            const float* A = (const float*)Aptr;
            const float4* src = (const float4*)(A + (size_t)(rowBase + stRow) * DM + k0 + stK);
            float4 f0 = src[0], f1 = src[1], f2 = src[2], f3 = src[3];
            uint4 p0, p1;
            p0.x = pack2(f0.x, f0.y); p0.y = pack2(f0.z, f0.w);
            p0.z = pack2(f1.x, f1.y); p0.w = pack2(f1.z, f1.w);
            p1.x = pack2(f2.x, f2.y); p1.y = pack2(f2.z, f2.w);
            p1.z = pack2(f3.x, f3.y); p1.w = pack2(f3.z, f3.w);
            *(uint4*)&As[stRow * 40 + stK]     = p0;
            *(uint4*)&As[stRow * 40 + stK + 8] = p1;
        }
        // ---- stage W tile (32 x 128) transposed -> Bs[col][k] ----
        {
            const float4* src = (const float4*)(W + (size_t)(k0 + wkRow) * DM + colBase + wnP);
            float4 f0 = src[0], f1 = src[1], f2 = src[2], f3 = src[3];
            float fv[16] = { f0.x, f0.y, f0.z, f0.w, f1.x, f1.y, f1.z, f1.w,
                             f2.x, f2.y, f2.z, f2.w, f3.x, f3.y, f3.z, f3.w };
            for (int j = 0; j < 16; ++j)
                Bs[(wnP + j) * 40 + wkRow] = bf16_of(fv[j]);
        }
#if USE_ASYNC_LDS
        if (ABF16) async_wait0();
#endif
        __syncthreads();
        // ---- WMMA: 4x2 tiles per wave ----
        for (int mi = 0; mi < 4; ++mi) {
            BFrag a;
            const int r  = wM * 64 + mi * 16 + (lane & 15);
            const int ka = (lane < 16) ? 0 : 8;
            a.q[0] = *(const uint4*)&As[r * 40 + ka];
            a.q[1] = *(const uint4*)&As[r * 40 + ka + 16];
            for (int ni = 0; ni < 2; ++ni) {
                BFrag b;
                const int c  = wN * 32 + ni * 16 + (lane & 15);
                const int kb = (lane < 16) ? 0 : 8;
                b.q[0] = *(const uint4*)&Bs[c * 40 + kb];
                b.q[1] = *(const uint4*)&Bs[c * 40 + kb + 16];
                acc[mi][ni] = wmma_bf16(a, b, acc[mi][ni]);
            }
        }
    }
    // ---- epilogue: bias + store ----
    for (int ni = 0; ni < 2; ++ni) {
        const int cAbs = colBase + wN * 32 + ni * 16 + (lane & 15);
        const float bv = bias[cAbs];
        for (int mi = 0; mi < 4; ++mi) {
            const int r0 = rowBase + wM * 64 + mi * 16 + ((lane < 16) ? 0 : 8);
            for (int i = 0; i < 8; ++i) {
                const float v = acc[mi][ni][i] + bv;
                if (OBF16) ((unsigned short*)Cptr)[(size_t)(r0 + i) * DM + cAbs] = bf16_of(v);
                else       ((float*)Cptr)[(size_t)(r0 + i) * DM + cAbs] = v;
            }
        }
    }
}

// ---------------------------------------------------------------------------
// FAVOR+ feature map.  data_dash = (x * 64^-1/4) @ projT  (per b,h: [128,64]x[64,256])
// MODE 0: query  -> row-max, exp, store q' (bf16)
// MODE 1: key p1 -> global max per (b,h) via atomicMax (no store)
// MODE 2: key p2 -> exp with global max, store k' (bf16)
// WG: 256 thr / 8 waves; wave owns 16 rows x all 256 cols (16 wmma tiles).
// ---------------------------------------------------------------------------
template<int MODE>
__global__ void __launch_bounds__(256)
feat_kernel(const unsigned short* __restrict__ QK,   // [B*L, 1024] bf16 (projected)
            const float* __restrict__ proj,          // [256, 64] f32
            unsigned short* __restrict__ outPrime,   // [B*L, H*M] bf16
            unsigned* __restrict__ gmax)             // [B*H]
{
    __shared__ __align__(16) unsigned short Ps[MMF * 72];  // projT as [n][k] == proj row-major
    __shared__ __align__(16) unsigned short As[128 * 72];  // scaled x tile [row][k]
    __shared__ float diagS[128];
    __shared__ float wredS[8];

    const int tid  = threadIdx.x;
    const int lane = tid & 31;
    const int wave = tid >> 5;
    const int lBase = blockIdx.x * 128;
    const int bh = blockIdx.y;
    const int b = bh >> 4, h = bh & 15;
    const float scale = 0.35355339059327373f;   // 64^{-1/4}
    const float ratio = 0.0625f;                // 1/sqrt(256)

    // stage proj (256x64 f32 -> bf16); Ps[n][k] = proj[n][k] directly
    {
        const int m = tid;
        const float4* src = (const float4*)(proj + m * DEPTHD);
        for (int kk = 0; kk < 16; ++kk) {
            float4 f = src[kk];
            uint2 t; t.x = pack2(f.x, f.y); t.y = pack2(f.z, f.w);
            *(uint2*)&Ps[m * 72 + kk * 4] = t;
        }
    }
    // stage scaled x tile (128 x 64 bf16)
    {
        const int row = tid >> 1;
        const int kp  = (tid & 1) * 32;
        const uint4* src = (const uint4*)(QK + (size_t)(b * LL + lBase + row) * DM + h * DEPTHD + kp);
        for (int q4 = 0; q4 < 4; ++q4) {
            uint4 uv = src[q4];
            unsigned w[4] = { uv.x, uv.y, uv.z, uv.w };
            for (int j = 0; j < 4; ++j) {
                float lo = f_of_bf16((unsigned short)(w[j] & 0xFFFFu)) * scale;
                float hi = f_of_bf16((unsigned short)(w[j] >> 16)) * scale;
                w[j] = pack2(lo, hi);
            }
            uint4 ov; ov.x = w[0]; ov.y = w[1]; ov.z = w[2]; ov.w = w[3];
            *(uint4*)&As[row * 72 + kp + q4 * 8] = ov;
        }
    }
    __syncthreads();
    if (tid < 128) {
        float s = 0.f;
        for (int k = 0; k < DEPTHD; ++k) { float v = f_of_bf16(As[tid * 72 + k]); s += v * v; }
        diagS[tid] = 0.5f * s;
    }
    __syncthreads();

    v8f acc[16] = {};
    for (int ks = 0; ks < DEPTHD; ks += 32) {
        BFrag a;
        const int r  = wave * 16 + (lane & 15);
        const int ka = ks + ((lane < 16) ? 0 : 8);
        a.q[0] = *(const uint4*)&As[r * 72 + ka];
        a.q[1] = *(const uint4*)&As[r * 72 + ka + 16];
        for (int ct = 0; ct < 16; ++ct) {
            BFrag bq;
            const int c  = ct * 16 + (lane & 15);
            const int kb = ks + ((lane < 16) ? 0 : 8);
            bq.q[0] = *(const uint4*)&Ps[c * 72 + kb];
            bq.q[1] = *(const uint4*)&Ps[c * 72 + kb + 16];
            acc[ct] = wmma_bf16(a, bq, acc[ct]);
        }
    }

    if (MODE == 1) {
        float pm = acc[0][0];
        for (int ct = 0; ct < 16; ++ct)
            for (int i = 0; i < 8; ++i) pm = fmaxf(pm, acc[ct][i]);
        for (int m = 1; m < 32; m <<= 1) pm = fmaxf(pm, __shfl_xor(pm, m, 32));
        if (lane == 0) wredS[wave] = pm;
        __syncthreads();
        if (tid == 0) {
            float g = wredS[0];
            for (int wv = 1; wv < 8; ++wv) g = fmaxf(g, wredS[wv]);
            atomicMax(&gmax[bh], enc_max(g));
        }
        return;
    }

    float rowmx[8];
    if (MODE == 0) {
        for (int i = 0; i < 8; ++i) rowmx[i] = acc[0][i];
        for (int ct = 1; ct < 16; ++ct)
            for (int i = 0; i < 8; ++i) rowmx[i] = fmaxf(rowmx[i], acc[ct][i]);
        // reduce over the 16 lanes of each half-wave (rows stay separated)
        for (int m = 1; m < 16; m <<= 1)
            for (int i = 0; i < 8; ++i) rowmx[i] = fmaxf(rowmx[i], __shfl_xor(rowmx[i], m, 32));
    } else { // MODE 2
        const float mx = dec_max(gmax[bh]);
        for (int i = 0; i < 8; ++i) rowmx[i] = mx;
    }

    const int rbase = wave * 16 + ((lane < 16) ? 0 : 8);
    for (int ct = 0; ct < 16; ++ct) {
        const int mcol = ct * 16 + (lane & 15);
        for (int i = 0; i < 8; ++i) {
            const int row = rbase + i;
            const float v = ratio * (__expf(acc[ct][i] - diagS[row] - rowmx[i]) + 1e-6f);
            const int l = lBase + row;
            outPrime[(size_t)((b * LL + l) * HH + h) * MMF + mcol] = bf16_of(v);
        }
    }
}

// ---------------------------------------------------------------------------
// kv[bh] = K'^T @ [V | 1]  per (b,h): [256, L] x [L, 80], split-K over L
// (column 64 of B is 1.0 so kv[:,64] = k_sum).  fp32 atomicAdd accumulation.
// WG: 256 thr / 8 waves; wave owns 2 row tiles x 5 col tiles; L-chunk = 512.
// ---------------------------------------------------------------------------
__global__ void __launch_bounds__(256)
kv_kernel(const unsigned short* __restrict__ Kp,  // [B*L, H*M] bf16
          const unsigned short* __restrict__ Vb,  // [B*L, 1024] bf16
          float* __restrict__ KV)                 // [B*H, 256, 80] f32
{
    __shared__ __align__(16) unsigned short As[MMF * 40];  // k'^T tile [m][l]
    __shared__ __align__(16) unsigned short Bs[NKV * 40];  // [n][l]

    const int tid  = threadIdx.x;
    const int lane = tid & 31;
    const int wave = tid >> 5;
    const int chunk = blockIdx.x;                  // 0..7
    const int bh = blockIdx.y;
    const int b = bh >> 4, h = bh & 15;
    const int lBase = chunk * 512;

    v8f acc[2][5] = {};

    // staging decomposition: 16 m's x 2 l-rows per thread (pack l-pairs -> b32)
    const int mb16 = (tid & 15) * 16;      // 0..240
    const int lp   = (tid >> 4) * 2;       // 0,2,..,30

    for (int l0 = 0; l0 < 512; l0 += 32) {
        if (l0 + 32 < 512) {
            __builtin_prefetch(Kp + (size_t)((b * LL + lBase + l0 + 32 + lp) * HH + h) * MMF + mb16, 0, 3);
        }
        __syncthreads();
        // stage A transposed: k'[32 l][256 m] -> As[m][l], dword-packed l-pairs
        {
            H16 r0, r1;
            const uint4* s0 = (const uint4*)(Kp +
                (size_t)((b * LL + lBase + l0 + lp) * HH + h) * MMF + mb16);
            const uint4* s1 = (const uint4*)(Kp +
                (size_t)((b * LL + lBase + l0 + lp + 1) * HH + h) * MMF + mb16);
            r0.q[0] = s0[0]; r0.q[1] = s0[1];
            r1.q[0] = s1[0]; r1.q[1] = s1[1];
            for (int j = 0; j < 16; ++j) {
                const unsigned d = (unsigned)r0.h[j] | ((unsigned)r1.h[j] << 16);
                *(unsigned*)&As[(mb16 + j) * 40 + lp] = d;
            }
        }
        // stage B: V columns + ones column + zero pad
        for (int idx = tid; idx < NKV * 32; idx += 256) {
            const int n = idx >> 5, lr = idx & 31;
            unsigned short val;
            if (n < 64)       val = Vb[(size_t)(b * LL + lBase + l0 + lr) * DM + h * DEPTHD + n];
            else if (n == 64) val = (unsigned short)0x3F80u;   // bf16(1.0)
            else              val = 0;
            Bs[n * 40 + lr] = val;
        }
        __syncthreads();
        for (int mi = 0; mi < 2; ++mi) {
            BFrag a;
            const int r  = (wave * 2 + mi) * 16 + (lane & 15);
            const int ka = (lane < 16) ? 0 : 8;
            a.q[0] = *(const uint4*)&As[r * 40 + ka];
            a.q[1] = *(const uint4*)&As[r * 40 + ka + 16];
            for (int ni = 0; ni < 5; ++ni) {
                BFrag bq;
                const int c  = ni * 16 + (lane & 15);
                const int kb = (lane < 16) ? 0 : 8;
                bq.q[0] = *(const uint4*)&Bs[c * 40 + kb];
                bq.q[1] = *(const uint4*)&Bs[c * 40 + kb + 16];
                acc[mi][ni] = wmma_bf16(a, bq, acc[mi][ni]);
            }
        }
    }
    for (int mi = 0; mi < 2; ++mi) {
        const int r0 = (wave * 2 + mi) * 16 + ((lane < 16) ? 0 : 8);
        for (int ni = 0; ni < 5; ++ni) {
            const int c = ni * 16 + (lane & 15);
            for (int i = 0; i < 8; ++i)
                atomicAdd(&KV[((size_t)bh * MMF + r0 + i) * NKV + c], acc[mi][ni][i]);
        }
    }
}

// ---------------------------------------------------------------------------
// av = Q' @ kv  per (b,h): [64, 256] x [256, 80]; col 64 = norm (q'.k_sum).
// attn = av[:, :64] / norm, written straight into the reference's
// transpose(0,2,1,3).reshape(B,-1,1024) layout so gemm_bias_kernel finishes.
// WG: 128 thr / 4 waves; wave owns one 16-row tile x 5 col tiles.
// ---------------------------------------------------------------------------
__global__ void __launch_bounds__(128)
av_kernel(const unsigned short* __restrict__ Qp,   // [B*L, H*M] bf16
          const float* __restrict__ KV,            // [B*H, 256, 80] f32
          unsigned short* __restrict__ attnF)      // [B, 4096, 1024] bf16 (flattened)
{
    __shared__ __align__(16) unsigned short As[64 * 40];
    __shared__ __align__(16) unsigned short Bs[NKV * 40];
    __shared__ float avS[64 * NKV];

    const int tid  = threadIdx.x;
    const int lane = tid & 31;
    const int wave = tid >> 5;
    const int lBase = blockIdx.x * 64;
    const int bh = blockIdx.y;
    const int b = bh >> 4, h = bh & 15;

    v8f acc[5] = {};

    for (int k0 = 0; k0 < MMF; k0 += 32) {
        __syncthreads();
        {   // stage A (64 x 32 bf16, contiguous in M) -- async copy if available
            const int row = tid >> 1;
            const int mp  = (tid & 1) * 16;
            const unsigned short* src = Qp +
                (size_t)((b * LL + lBase + row) * HH + h) * MMF + k0 + mp;
#if USE_ASYNC_LDS
            async_copy16(src,     &As[row * 40 + mp]);
            async_copy16(src + 8, &As[row * 40 + mp + 8]);
#else
            *(uint4*)&As[row * 40 + mp]     = ((const uint4*)src)[0];
            *(uint4*)&As[row * 40 + mp + 8] = ((const uint4*)src)[1];
#endif
        }
        // stage B: kv rows k0..k0+31 transposed -> Bs[n][k], f32 -> bf16
        for (int idx = tid; idx < NKV * 32; idx += 128) {
            const int n = idx >> 5, kr = idx & 31;
            Bs[n * 40 + kr] = bf16_of(KV[((size_t)bh * MMF + k0 + kr) * NKV + n]);
        }
#if USE_ASYNC_LDS
        async_wait0();
#endif
        __syncthreads();
        BFrag a;
        const int r  = wave * 16 + (lane & 15);
        const int ka = (lane < 16) ? 0 : 8;
        a.q[0] = *(const uint4*)&As[r * 40 + ka];
        a.q[1] = *(const uint4*)&As[r * 40 + ka + 16];
        for (int ni = 0; ni < 5; ++ni) {
            BFrag bq;
            const int c  = ni * 16 + (lane & 15);
            const int kb = (lane < 16) ? 0 : 8;
            bq.q[0] = *(const uint4*)&Bs[c * 40 + kb];
            bq.q[1] = *(const uint4*)&Bs[c * 40 + kb + 16];
            acc[ni] = wmma_bf16(a, bq, acc[ni]);
        }
    }
    {   // spill av to LDS for cross-lane norm access
        const int r0 = wave * 16 + ((lane < 16) ? 0 : 8);
        for (int ni = 0; ni < 5; ++ni) {
            const int c = ni * 16 + (lane & 15);
            for (int i = 0; i < 8; ++i) avS[(r0 + i) * NKV + c] = acc[ni][i];
        }
    }
    __syncthreads();
    {
        const int row = tid >> 1;
        const int dh  = (tid & 1) * 32;
        const float inv = 1.f / avS[row * NKV + 64];
        const int l = lBase + row;
        const int rF = h * 256 + (l >> 4);          // faithful reshape row
        const int cB = (l & 15) * 64;               // faithful reshape col base
        for (int j = 0; j < 32; ++j) {
            const float v = avS[row * NKV + dh + j] * inv;
            attnF[((size_t)b * 4096 + rF) * DM + cB + dh + j] = bf16_of(v);
        }
    }
}

// ---------------------------------------------------------------------------
extern "C" void kernel_launch(void* const* d_in, const int* in_sizes, int n_in,
                              void* d_out, int out_size, void* d_ws, size_t ws_size,
                              hipStream_t stream)
{
    (void)in_sizes; (void)n_in; (void)out_size; (void)ws_size;
    const float* query = (const float*)d_in[0];
    const float* key_  = (const float*)d_in[1];
    const float* value = (const float*)d_in[2];
    const float* Wq = (const float*)d_in[3];
    const float* bq = (const float*)d_in[4];
    const float* Wk = (const float*)d_in[5];
    const float* bk = (const float*)d_in[6];
    const float* Wv = (const float*)d_in[7];
    const float* bv = (const float*)d_in[8];
    const float* Wo = (const float*)d_in[9];
    const float* bo = (const float*)d_in[10];
    const float* proj = (const float*)d_in[11];

    char* ws = (char*)d_ws;
    size_t off = 0;
    auto take = [&](size_t bytes) -> void* {
        void* p = ws + off;
        off += (bytes + 255) & ~(size_t)255;
        return p;
    };
    unsigned short* qb = (unsigned short*)take((size_t)NROWS * DM * 2);
    unsigned short* kb = (unsigned short*)take((size_t)NROWS * DM * 2);
    unsigned short* vb = (unsigned short*)take((size_t)NROWS * DM * 2);
    unsigned short* qp = (unsigned short*)take((size_t)NROWS * HH * MMF * 2);
    unsigned short* kp = (unsigned short*)take((size_t)NROWS * HH * MMF * 2);
    float*          kv = (float*)take((size_t)BB * HH * MMF * NKV * 4);
    unsigned*     gmax = (unsigned*)take((size_t)BB * HH * 4);
    unsigned short* af = (unsigned short*)take((size_t)BB * 4096 * DM * 2);

    const dim3 gGemm(NROWS / 128, DM / 128);
    const dim3 gFeat(LL / 128, BB * HH);
    const dim3 gKv(8, BB * HH);
    const dim3 gAv(LL / 64, BB * HH);

    // 1) projections (fp32 in, bf16 out)
    gemm_bias_kernel<false, true><<<gGemm, 256, 0, stream>>>(query, Wq, bq, qb);
    gemm_bias_kernel<false, true><<<gGemm, 256, 0, stream>>>(key_,  Wk, bk, kb);
    gemm_bias_kernel<false, true><<<gGemm, 256, 0, stream>>>(value, Wv, bv, vb);

    // 2) feature maps
    feat_kernel<0><<<gFeat, 256, 0, stream>>>(qb, proj, qp, gmax);
    (void)hipMemsetAsync(gmax, 0, (size_t)BB * HH * 4, stream);
    feat_kernel<1><<<gFeat, 256, 0, stream>>>(kb, proj, kp, gmax);
    feat_kernel<2><<<gFeat, 256, 0, stream>>>(kb, proj, kp, gmax);

    // 3) kv (+ k_sum via ones column), split-K atomics
    (void)hipMemsetAsync(kv, 0, (size_t)BB * HH * MMF * NKV * 4, stream);
    kv_kernel<<<gKv, 256, 0, stream>>>(kp, vb, kv);

    // 4) av + normalization, directly into flattened layout
    av_kernel<<<gAv, 128, 0, stream>>>(qp, kv, af);

    // 5) output projection (bf16 in, fp32 out)
    gemm_bias_kernel<true, false><<<gGemm, 256, 0, stream>>>(af, Wo, bo, (float*)d_out);
}
